// MultiScaleRetention_38173669327186
// MI455X (gfx1250) — matrix operand order repaired and checked by reading, not compiled, not run on hardware
//
#include <hip/hip_runtime.h>

typedef __attribute__((ext_vector_type(16))) __bf16 v16bf;
typedef __attribute__((ext_vector_type(8)))  __bf16 bf16x8;
typedef __attribute__((ext_vector_type(8)))  float  v8f;
typedef __attribute__((ext_vector_type(4)))  float  f32x4;
typedef __attribute__((ext_vector_type(4)))  unsigned int u32x4;
typedef __attribute__((ext_vector_type(8)))  unsigned int u32x8;

#define BB 8
#define SS 2048
#define EE 512
#define HH 8
#define HSD 64
#define ST 512  // S / N_AGENTS

// ---------------------------------------------------------------------------
// WMMA fragment helpers (wave32 16x16x32 bf16, this file's contiguous k order, same for A and B)
// A lane(hf=lane>>4): element p at K = hf*16 + p
//   -> two contiguous 16B runs.
// B lane: element p at K = hf*16 + p, N = lane&15 -> one contiguous 32B run.
// C/D: row r -> M = r + 8*hf, N = lane&15.
// ---------------------------------------------------------------------------
__device__ __forceinline__ v16bf load_a_bf(const __bf16* base, int hf) {
  bf16x8 lo = *(const bf16x8*)(base + hf * 16);
  bf16x8 hi = *(const bf16x8*)(base + hf * 16 + 8);
  return __builtin_shufflevector(lo, hi, 0, 1, 2, 3, 4, 5, 6, 7, 8, 9, 10, 11,
                                 12, 13, 14, 15);
}
__device__ __forceinline__ v16bf load_b_bf(const __bf16* base, int hf) {
  const bf16x8* p = (const bf16x8*)(base + hf * 16);
  bf16x8 lo = p[0], hi = p[1];
  return __builtin_shufflevector(lo, hi, 0, 1, 2, 3, 4, 5, 6, 7, 8, 9, 10, 11,
                                 12, 13, 14, 15);
}
__device__ __forceinline__ v16bf load_a_f32(const float* base, int hf) {
  f32x4 x0 = *(const f32x4*)(base + hf * 16);
  f32x4 x1 = *(const f32x4*)(base + hf * 16 + 4);
  f32x4 x2 = *(const f32x4*)(base + hf * 16 + 8);
  f32x4 x3 = *(const f32x4*)(base + hf * 16 + 12);
  v16bf r;
#pragma unroll
  for (int i = 0; i < 4; ++i) {
    r[i] = (__bf16)x0[i];
    r[4 + i] = (__bf16)x1[i];
    r[8 + i] = (__bf16)x2[i];
    r[12 + i] = (__bf16)x3[i];
  }
  return r;
}
__device__ __forceinline__ v8f wmma_bf16(v16bf a, v16bf b, v8f c) {
  return __builtin_amdgcn_wmma_f32_16x16x32_bf16(false, a, false, b, (short)0, c,
                                                 false, false);
}
__device__ __forceinline__ float kappa_for_head(int h) {
  const float a = -3.4657359027997265f;  // log(1/32)
  const float b = -6.2383246250395077f;  // log(1/512)
  return 1.0f - __expf(a + (b - a) * ((float)h / 7.0f));
}

// ---------------------------------------------------------------------------
// Tensor Data Mover: 2D tile load Global -> LDS (CDNA5 ISA)
// D# group0: count=1, lds_addr, global_addr(57b), type=2.  group1: data_size=2B,
// tensor_dim0/1, tile_dim0/1, tensor_dim0_stride.  groups 2/3 zero (<=2D).
// ---------------------------------------------------------------------------
__device__ __forceinline__ void tdm_load_2d_bf16(unsigned lds_off,
                                                 const void* gaddr,
                                                 unsigned tile_d0, unsigned tile_d1,
                                                 unsigned tens_d0, unsigned tens_d1,
                                                 unsigned stride0) {
  unsigned long long ga = (unsigned long long)gaddr;
  u32x4 g0;
  g0[0] = 1u;                                     // count=1, user descriptor
  g0[1] = lds_off;                                // LDS byte address
  g0[2] = (unsigned)ga;                           // global addr [31:0]
  g0[3] = (unsigned)((ga >> 32) & 0x01FFFFFFull)  // global addr [56:32]
          | (2u << 30);                           // type=2 ("image")
  u32x8 g1;
  g1[0] = (1u << 16);                             // data_size=1 -> 2 bytes
  g1[1] = (tens_d0 & 0xFFFFu) << 16;              // tensor_dim0[15:0] @bit48
  g1[2] = ((tens_d0 >> 16) & 0xFFFFu) | ((tens_d1 & 0xFFFFu) << 16);
  g1[3] = ((tens_d1 >> 16) & 0xFFFFu) | (tile_d0 << 16);  // tile_dim0 @bit112
  g1[4] = tile_d1 & 0xFFFFu;                      // tile_dim1 (tile_dim2=0)
  g1[5] = stride0;                                // tensor_dim0_stride[31:0]
  g1[6] = 0;                                      // stride hi, dim1_stride lo
  g1[7] = 0;
  u32x4 gz = {0u, 0u, 0u, 0u};                    // groups 2/3 (2D tensor)
  asm volatile("tensor_load_to_lds %0, %1, %2, %3"
               :
               : "s"(g0), "s"(g1), "s"(gz), "s"(gz)
               : "memory");
}

// ---------------------------------------------------------------------------
// Kernel: per-batch cumsum of timestep dones, first-done index, any-done
// ---------------------------------------------------------------------------
__global__ void cum_kernel(const unsigned char* __restrict__ dones,
                           int* __restrict__ cum, int* __restrict__ firsts,
                           int* __restrict__ anyd) {
  int b = blockIdx.x;
  if (threadIdx.x == 0) {
    int run = 0, first = ST;
    for (int t = 0; t < ST; ++t) {
      int d = dones[b * SS + t * 4] ? 1 : 0;
      if (d && first == ST) first = t;
      run += d;
      cum[b * ST + t] = run;  // inclusive cumsum, matches jnp.cumsum
    }
    firsts[b] = first;
    anyd[b] = (run > 0) ? 1 : 0;
  }
}

// ---------------------------------------------------------------------------
// Kernel: pack weights transposed to bf16: Wt[n*512 + k]
// wmode 0: W is (H,E,hs): W[(n>>6)*E*hs + k*hs + (n&63)]; wmode 1: W[k*E + n]
// ---------------------------------------------------------------------------
__global__ __launch_bounds__(256) void pack_w(const float* __restrict__ W,
                                              __bf16* __restrict__ Wt, int wmode) {
  int idx = blockIdx.x * 256 + threadIdx.x;  // 0 .. 512*512-1
  int n = idx >> 9, k = idx & 511;
  float v = (wmode == 0) ? W[(n >> 6) * (EE * HSD) + k * HSD + (n & 63)]
                         : W[k * EE + n];
  Wt[idx] = (__bf16)v;
}

// ---------------------------------------------------------------------------
// Kernel: pack hstate transposed to bf16: Ht[(g*64+v)*64+d] = hstate[(g*64+d)*64+v]
// ---------------------------------------------------------------------------
__global__ __launch_bounds__(256) void pack_h(const float* __restrict__ Hs,
                                              __bf16* __restrict__ Ht) {
  int idx = blockIdx.x * 256 + threadIdx.x;  // 0 .. 8*8*64*64-1
  int g = idx >> 12;
  int v = (idx >> 6) & 63, d = idx & 63;
  Ht[idx] = (__bf16)Hs[(g * 64 + d) * 64 + v];
}

// ---------------------------------------------------------------------------
// Kernel: projection GEMM  acc = X_f32[M,512] @ Wt^T  (Wt pre-packed bf16)
// Optionally writes row-major C[m*512+n] and/or transposed Ct[((b*8+h)*64+d)*2048+s]
// ---------------------------------------------------------------------------
__global__ __launch_bounds__(128) void gemm_proj(const float* __restrict__ X,
                                                 const __bf16* __restrict__ Wt,
                                                 __bf16* __restrict__ C,
                                                 __bf16* __restrict__ Ct) {
  int lane = threadIdx.x & 31, wave = threadIdx.x >> 5, hf = lane >> 4;
  int m0 = (blockIdx.y * 4 + wave) * 16;
  int n0 = blockIdx.x * 16;
  int mrow = m0 + (lane & 15);
  int ncol = n0 + (lane & 15);
  const float* xrow = X + (size_t)mrow * EE;
  const __bf16* wrow = Wt + (size_t)ncol * EE;
  v8f acc = {};
  for (int k0 = 0; k0 < EE; k0 += 32) {
    __builtin_prefetch((const void*)(xrow + k0 + 128), 0, 0);
    acc = wmma_bf16(load_a_f32(xrow + k0, hf), load_b_bf(wrow + k0, hf), acc);
  }
  if (C) {
#pragma unroll
    for (int r = 0; r < 8; ++r) {
      int m = m0 + r + hf * 8;
      C[(size_t)m * EE + ncol] = (__bf16)acc[r];
    }
  }
  if (Ct) {
    int hh = ncol >> 6, dd = ncol & 63;
#pragma unroll
    for (int r = 0; r < 8; ++r) {
      int m = m0 + r + hf * 8;
      int bb = m >> 11, s = m & 2047;
      Ct[((size_t)(bb * HH + hh) * HSD + dd) * SS + s] = (__bf16)acc[r];
    }
  }
}

// ---------------------------------------------------------------------------
// Kernel: retention core.  One wave per (b, h, 16-row tile).
// Y = ((qp kp^T) * dm) vp + (qp @ hstate) * e ; dm from cum[] + LDS kappa LUT.
// V tiles are TDM-prefetched into double-buffered LDS (s_wait_tensorcnt 1),
// overlapped with QK^T WMMAs; consumed via ds_load_b128.
// ---------------------------------------------------------------------------
__global__ __launch_bounds__(32) void retention_kernel(
    const __bf16* __restrict__ Q, const __bf16* __restrict__ K,
    const __bf16* __restrict__ Vt, const __bf16* __restrict__ Ht,
    const int* __restrict__ cum, const int* __restrict__ firsts,
    float* __restrict__ Y) {
  __shared__ __bf16 sP[16 * 32];      // masked P tile, rows x keys
  __shared__ __bf16 sV[2][64 * 32];   // double-buffered V tile: [col][key]
  __shared__ float sKpow[ST];         // kappa^d LUT
  int lane = threadIdx.x & 31, hf = lane >> 4;
  int it = blockIdx.x, h = blockIdx.y, b = blockIdx.z;
  float logk = __logf(kappa_for_head(h));
  int m0 = it * 16;
  const int* cumb = cum + b * ST;

  // fill decay LUT (one wave): kappa^d for d = 0..511
  for (int i = lane; i < ST; i += 32) sKpow[i] = __expf(logk * (float)i);

  // issue first V-tile TDM load so it overlaps the cross-term WMMAs
  const __bf16* vbase = Vt + (size_t)(b * HH + h) * HSD * SS;  // +col*SS+key
  unsigned sv_off0 = (unsigned)(unsigned long long)(void*)&sV[0][0];
  unsigned sv_off1 = (unsigned)(unsigned long long)(void*)&sV[1][0];
  int nblk = m0 / 32 + 1;
  tdm_load_2d_bf16(sv_off0, (const void*)vbase,
                   /*tile_d0=*/32, /*tile_d1=*/64,
                   /*tens_d0=*/SS, /*tens_d1=*/HSD, /*stride0=*/SS);

  const __bf16* qrow = Q + ((size_t)(b * SS + m0 + (lane & 15))) * EE + h * HSD;
  v16bf aQ0 = load_a_bf(qrow, hf);       // hs dims 0..31
  v16bf aQ1 = load_a_bf(qrow + 32, hf);  // hs dims 32..63

  // ---- cross = (qp @ hstate[b,h]) * e ----
  v8f acc[4];
  {
    int firstb = firsts[b];
#pragma unroll
    for (int ct = 0; ct < 4; ++ct) {
      const __bf16* hb =
          Ht + ((size_t)(b * HH + h) * HSD + ct * 16 + (lane & 15)) * HSD;
      v8f ce = {};
      ce = wmma_bf16(aQ0, load_b_bf(hb, hf), ce);
      ce = wmma_bf16(aQ1, load_b_bf(hb + 32, hf), ce);
#pragma unroll
      for (int r = 0; r < 8; ++r) {
        int t = (m0 + r + hf * 8) >> 2;
        float e = (t < firstb) ? __expf(logk * (float)(t + 1)) : 0.0f;
        acc[ct][r] = ce[r] * e;
      }
    }
  }
  __syncthreads();  // sKpow visible

  // ---- inner: loop over 32-key blocks, double-buffered V ----
  for (int blk = 0; blk < nblk; ++blk) {
    if (blk + 1 < nblk) {  // prefetch next tile into the other buffer
      tdm_load_2d_bf16((blk & 1) ? sv_off0 : sv_off1,
                       (const void*)(vbase + (blk + 1) * 32),
                       32, 64, SS, HSD, SS);
    }
#pragma unroll
    for (int hv = 0; hv < 2; ++hv) {  // two 16-key halves of the block
      int j0 = blk * 32 + hv * 16;
      const __bf16* krow =
          K + ((size_t)(b * SS + j0 + (lane & 15))) * EE + h * HSD;
      v8f s = {};
      s = wmma_bf16(aQ0, load_b_bf(krow, hf), s);
      s = wmma_bf16(aQ1, load_b_bf(krow + 32, hf), s);
      // decay mask dm on the fly (LUT), stage bf16 P into LDS (16x32)
#pragma unroll
      for (int r = 0; r < 8; ++r) {
        int m = r + hf * 8;
        int ig = m0 + m;
        int jg = j0 + (lane & 15);
        int ti = ig >> 2, tj = jg >> 2;
        bool ok = (ig >= jg) && ((ti == tj) || (cumb[ti] == cumb[tj]));
        float w = ok ? sKpow[ti - tj] : 0.0f;
        sP[m * 32 + hv * 16 + (lane & 15)] = (__bf16)(s[r] * w);
      }
    }
    __syncthreads();
    if (blk + 1 < nblk) __builtin_amdgcn_s_wait_tensorcnt(1);
    else                __builtin_amdgcn_s_wait_tensorcnt(0);
    const __bf16* sVb = &sV[blk & 1][0];
    v16bf aP = load_a_bf(&sP[(lane & 15) * 32], hf);
#pragma unroll
    for (int ct = 0; ct < 4; ++ct) {
      acc[ct] = wmma_bf16(aP, load_b_bf(sVb + (ct * 16 + (lane & 15)) * 32, hf),
                          acc[ct]);
    }
    __syncthreads();
  }

#pragma unroll
  for (int ct = 0; ct < 4; ++ct)
#pragma unroll
    for (int r = 0; r < 8; ++r) {
      int m = m0 + r + hf * 8;
      Y[((size_t)(b * SS + m)) * EE + h * HSD + ct * 16 + (lane & 15)] =
          acc[ct][r];
    }
}

// ---------------------------------------------------------------------------
// Kernel: h_next[b,h] = kp^T @ (vp * w_last) + hstate * kappa^St * delta
// w_last precomputed into LDS once per block; Kt/Vt give contiguous b128 frags.
// ---------------------------------------------------------------------------
__global__ __launch_bounds__(256) void nexth_kernel(
    const __bf16* __restrict__ Kt, const __bf16* __restrict__ Vt,
    const float* __restrict__ hstate, const int* __restrict__ cum,
    const int* __restrict__ anyd, float* __restrict__ out) {
  __shared__ float sW[SS];  // w_last[s] = dm[b, S-1, s]
  int lane = threadIdx.x & 31, wave = threadIdx.x >> 5, hf = lane >> 4;
  int b = blockIdx.x >> 3, h = blockIdx.x & 7;
  float logk = __logf(kappa_for_head(h));
  const int* cumb = cum + b * ST;
  int cumLast = cumb[ST - 1];
  for (int s = threadIdx.x; s < SS; s += 256) {
    int tj = s >> 2;
    bool ok = (tj == ST - 1) || (cumb[tj] == cumLast);
    sW[s] = ok ? __expf(logk * (float)(ST - 1 - tj)) : 0.0f;
  }
  __syncthreads();

  int mt0 = wave >> 2, nt0 = wave & 3;  // tile (d-block, v-block)
  int mt1 = (wave + 8) >> 2, nt1 = (wave + 8) & 3;
  const __bf16* ka0 =
      Kt + ((size_t)(b * HH + h) * HSD + mt0 * 16 + (lane & 15)) * SS;
  const __bf16* ka1 =
      Kt + ((size_t)(b * HH + h) * HSD + mt1 * 16 + (lane & 15)) * SS;
  const __bf16* vb0 =
      Vt + ((size_t)(b * HH + h) * HSD + nt0 * 16 + (lane & 15)) * SS;
  const __bf16* vb1 =
      Vt + ((size_t)(b * HH + h) * HSD + nt1 * 16 + (lane & 15)) * SS;
  v8f acc0 = {}, acc1 = {};
  for (int s0 = 0; s0 < SS; s0 += 32) {
    const float* wp = &sW[s0 + hf * 16];
    v16bf bv0 = load_b_bf(vb0 + s0, hf);
    v16bf bv1 = load_b_bf(vb1 + s0, hf);
    v16bf bs0, bs1;
#pragma unroll
    for (int p = 0; p < 16; ++p) {
      float w = wp[p];
      bs0[p] = (__bf16)((float)bv0[p] * w);
      bs1[p] = (__bf16)((float)bv1[p] * w);
    }
    acc0 = wmma_bf16(load_a_bf(ka0 + s0, hf), bs0, acc0);
    acc1 = wmma_bf16(load_a_bf(ka1 + s0, hf), bs1, acc1);
  }
  float cd = __expf(logk * (float)ST);
  float delta = anyd[b] ? 0.0f : 1.0f;
#pragma unroll
  for (int r = 0; r < 8; ++r) {
    int d0 = mt0 * 16 + r + hf * 8, v0 = nt0 * 16 + (lane & 15);
    int i0 = ((b * HH + h) * HSD + d0) * HSD + v0;
    out[i0] = acc0[r] + hstate[i0] * cd * delta;
    int d1 = mt1 * 16 + r + hf * 8, v1 = nt1 * 16 + (lane & 15);
    int i1 = ((b * HH + h) * HSD + d1) * HSD + v1;
    out[i1] = acc1[r] + hstate[i1] * cd * delta;
  }
}

// ---------------------------------------------------------------------------
// Kernel: group-norm over hs=64 per (token, head); one wave per group.
// ---------------------------------------------------------------------------
__global__ __launch_bounds__(256) void gnorm_kernel(
    const float* __restrict__ Y, const float* __restrict__ gn_scale,
    const float* __restrict__ gn_bias, __bf16* __restrict__ Yn) {
  int lane = threadIdx.x & 31, wave = threadIdx.x >> 5;
  int grp = blockIdx.x * 8 + wave;
  int m = grp >> 3, h = grp & 7;
  size_t base = (size_t)m * EE + h * HSD;
  float v0 = Y[base + lane * 2], v1 = Y[base + lane * 2 + 1];
  float s = v0 + v1, q = v0 * v0 + v1 * v1;
#pragma unroll
  for (int off = 1; off < 32; off <<= 1) {
    s += __shfl_xor(s, off, 32);
    q += __shfl_xor(q, off, 32);
  }
  float mu = s * (1.0f / 64.0f);
  float var = q * (1.0f / 64.0f) - mu * mu;
  float rs = rsqrtf(var + 1e-6f);
  int c0 = h * HSD + lane * 2;
  Yn[base + lane * 2]     = (__bf16)((v0 - mu) * rs * gn_scale[c0] + gn_bias[c0]);
  Yn[base + lane * 2 + 1] = (__bf16)((v1 - mu) * rs * gn_scale[c0 + 1] + gn_bias[c0 + 1]);
}

// ---------------------------------------------------------------------------
// Kernel: Gz = swish(key @ W_G)  (packed Wt, f32 out)
// ---------------------------------------------------------------------------
__global__ __launch_bounds__(128) void gemm_gate(const float* __restrict__ X,
                                                 const __bf16* __restrict__ Wt,
                                                 float* __restrict__ G) {
  int lane = threadIdx.x & 31, wave = threadIdx.x >> 5, hf = lane >> 4;
  int m0 = (blockIdx.y * 4 + wave) * 16;
  int n0 = blockIdx.x * 16;
  int mrow = m0 + (lane & 15);
  int ncol = n0 + (lane & 15);
  const float* xrow = X + (size_t)mrow * EE;
  const __bf16* wrow = Wt + (size_t)ncol * EE;
  v8f acc = {};
  for (int k0 = 0; k0 < EE; k0 += 32) {
    __builtin_prefetch((const void*)(xrow + k0 + 128), 0, 0);
    acc = wmma_bf16(load_a_f32(xrow + k0, hf), load_b_bf(wrow + k0, hf), acc);
  }
#pragma unroll
  for (int r = 0; r < 8; ++r) {
    int m = m0 + r + hf * 8;
    float x = acc[r];
    G[(size_t)m * EE + ncol] = x / (1.0f + __expf(-x));  // swish
  }
}

// ---------------------------------------------------------------------------
// Kernel: out = Gz * (Yn @ W_O)  (Yn bf16 A-frags, packed Wt, f32 out)
// ---------------------------------------------------------------------------
__global__ __launch_bounds__(128) void gemm_out(const __bf16* __restrict__ Yn,
                                                const __bf16* __restrict__ Wt,
                                                const float* __restrict__ G,
                                                float* __restrict__ Out) {
  int lane = threadIdx.x & 31, wave = threadIdx.x >> 5, hf = lane >> 4;
  int m0 = (blockIdx.y * 4 + wave) * 16;
  int n0 = blockIdx.x * 16;
  int mrow = m0 + (lane & 15);
  int ncol = n0 + (lane & 15);
  const __bf16* yrow = Yn + (size_t)mrow * EE;
  const __bf16* wrow = Wt + (size_t)ncol * EE;
  v8f acc = {};
  for (int k0 = 0; k0 < EE; k0 += 32) {
    __builtin_prefetch((const void*)(yrow + k0 + 128), 0, 0);
    acc = wmma_bf16(load_a_bf(yrow + k0, hf), load_b_bf(wrow + k0, hf), acc);
  }
#pragma unroll
  for (int r = 0; r < 8; ++r) {
    int m = m0 + r + hf * 8;
    Out[(size_t)m * EE + ncol] = G[(size_t)m * EE + ncol] * acc[r];
  }
}

// ---------------------------------------------------------------------------
extern "C" void kernel_launch(void* const* d_in, const int* in_sizes, int n_in,
                              void* d_out, int out_size, void* d_ws, size_t ws_size,
                              hipStream_t stream) {
  const float* key    = (const float*)d_in[0];
  const float* query  = (const float*)d_in[1];
  const float* value  = (const float*)d_in[2];
  const float* hstate = (const float*)d_in[3];
  const unsigned char* dones = (const unsigned char*)d_in[4];
  const float* W_Q = (const float*)d_in[5];
  const float* W_K = (const float*)d_in[6];
  const float* W_V = (const float*)d_in[7];
  const float* W_G = (const float*)d_in[8];
  const float* W_O = (const float*)d_in[9];
  const float* gn_scale = (const float*)d_in[10];
  const float* gn_bias  = (const float*)d_in[11];
  float* out = (float*)d_out;

  const size_t M = (size_t)BB * SS;  // 16384 rows
  char* ws = (char*)d_ws;
  size_t off = 0;
  auto alloc = [&](size_t bytes) {
    void* p = ws + off;
    off = (off + bytes + 255) & ~(size_t)255;
    return p;
  };
  __bf16* Qall = (__bf16*)alloc(M * EE * 2);
  __bf16* Kall = (__bf16*)alloc(M * EE * 2);
  __bf16* Kt   = (__bf16*)alloc(M * EE * 2);
  __bf16* Vt   = (__bf16*)alloc(M * EE * 2);
  float*  Y    = (float*)alloc(M * EE * 4);
  __bf16* Yn   = (__bf16*)alloc(M * EE * 2);
  float*  Gz   = (float*)alloc(M * EE * 4);
  __bf16* WtQ  = (__bf16*)alloc((size_t)EE * EE * 2);
  __bf16* WtK  = (__bf16*)alloc((size_t)EE * EE * 2);
  __bf16* WtV  = (__bf16*)alloc((size_t)EE * EE * 2);
  __bf16* WtG  = (__bf16*)alloc((size_t)EE * EE * 2);
  __bf16* WtO  = (__bf16*)alloc((size_t)EE * EE * 2);
  __bf16* Ht   = (__bf16*)alloc((size_t)BB * HH * HSD * HSD * 2);
  int*    cum  = (int*)alloc((size_t)BB * ST * 4);
  int*    firsts = (int*)alloc(64);
  int*    anyd   = (int*)alloc(64);

  cum_kernel<<<BB, 32, 0, stream>>>(dones, cum, firsts, anyd);
  pack_w<<<1024, 256, 0, stream>>>(W_Q, WtQ, 0);
  pack_w<<<1024, 256, 0, stream>>>(W_K, WtK, 0);
  pack_w<<<1024, 256, 0, stream>>>(W_V, WtV, 0);
  pack_w<<<1024, 256, 0, stream>>>(W_G, WtG, 1);
  pack_w<<<1024, 256, 0, stream>>>(W_O, WtO, 1);
  pack_h<<<1024, 256, 0, stream>>>(hstate, Ht);

  dim3 gg(EE / 16, (unsigned)(M / 64));  // 32 x 256 blocks, 4 waves each
  gemm_proj<<<gg, 128, 0, stream>>>(query, WtQ, Qall, nullptr);
  gemm_proj<<<gg, 128, 0, stream>>>(key,   WtK, Kall, Kt);
  gemm_proj<<<gg, 128, 0, stream>>>(value, WtV, nullptr, Vt);

  retention_kernel<<<dim3(SS / 16, HH, BB), 32, 0, stream>>>(
      Qall, Kall, Vt, Ht, cum, firsts, Y);

  nexth_kernel<<<BB * HH, 256, 0, stream>>>(Kt, Vt, hstate, cum, anyd,
                                            out + M * EE);

  gnorm_kernel<<<(unsigned)(M * HH / 8), 256, 0, stream>>>(Y, gn_scale, gn_bias, Yn);
  gemm_gate<<<gg, 128, 0, stream>>>(key, WtG, Gz);
  gemm_out<<<gg, 128, 0, stream>>>(Yn, WtO, Gz, out);
}
